// SelfMultiheadAttn_33285996544142
// MI455X (gfx1250) — compile-verified, matrix operand order checked by reading, not run on hardware
//
#include <hip/hip_runtime.h>
#include <hip/hip_bf16.h>
#include <stdint.h>

#define T_DIM 1024
#define B_DIM 16
#define D_DIM 1024
#define H_DIM 16
#define HD    64
#define NROW  (T_DIM * B_DIM)   /* 16384 rows of x */
#define E3D   (3 * D_DIM)       /* 3072 */

typedef __bf16 bf16_t;
typedef __attribute__((ext_vector_type(16))) __bf16        v16bf;
typedef __attribute__((ext_vector_type(8)))  float         v8f;
typedef __attribute__((ext_vector_type(4)))  unsigned int  u32x4;
typedef __attribute__((ext_vector_type(4)))  int           i32x4;
typedef __attribute__((ext_vector_type(8)))  int           i32x8;

union ABReg  { v16bf v; u32x4 q[2]; uint32_t u[8]; };
union AccReg { v8f v; float f[8]; };

__device__ __forceinline__ uint32_t pack2_bf16(float lo, float hi) {
    union { __bf16 b[2]; uint32_t u; } p;
    p.b[0] = (__bf16)lo; p.b[1] = (__bf16)hi;
    return p.u;
}

// TDM: issue a 2-D tile load (bf16 elements) global -> LDS via the Tensor Data
// Mover. Descriptor bitfields per CDNA5 ISA §8 (D# groups 0/1; groups 2/3 zero
// for 2-D tiles). This toolchain uses the 6-arg builtin form:
//   (u32x4 g0, i32x8 g1, i32x4 g2, i32x4 g3, i32x8 extra, i32 cpol)
__device__ __forceinline__ void tdm_load_tile_2d(
    uint32_t lds_off, const bf16_t* gptr,
    uint32_t tile_d0, uint32_t tile_d1,
    uint32_t tensor_d0, uint32_t tensor_d1, uint32_t stride0)
{
    const uint64_t ga = (uint64_t)(uintptr_t)gptr;
    u32x4 g0;
    g0[0] = 1u;                                   // count=1, user descriptor
    g0[1] = lds_off;                              // lds_addr (bytes)
    g0[2] = (uint32_t)ga;                         // global_addr[31:0]
    g0[3] = (uint32_t)(ga >> 32) | (2u << 30);    // global_addr[56:32] | type=2
    i32x8 g1;
    g1[0] = 0x00010000;                           // wg_mask=0, data_size=2B
    g1[1] = (int)(tensor_d0 << 16);               // tensor_dim0[15:0]
    g1[2] = (int)((tensor_d0 >> 16) | (tensor_d1 << 16));
    g1[3] = (int)((tensor_d1 >> 16) | (tile_d0 << 16));
    g1[4] = (int)tile_d1;                         // tile_dim1, tile_dim2=0
    g1[5] = (int)stride0;                         // tensor_dim0_stride[31:0]
    g1[6] = 0;
    g1[7] = 0;
    i32x4 gz4 = {0, 0, 0, 0};
    i32x8 gz8 = {0, 0, 0, 0, 0, 0, 0, 0};
    __builtin_amdgcn_tensor_load_to_lds(g0, g1, gz4, gz4, gz8, 0);
}

// ---------------------------------------------------------------- convert
__global__ void cvt_f32_bf16(const float* __restrict__ in, bf16_t* __restrict__ out, int n) {
    int i = blockIdx.x * blockDim.x + threadIdx.x;
    int stride = gridDim.x * blockDim.x;
    for (; i < n; i += stride) out[i] = (bf16_t)in[i];
}

// ---------------------------------------------------------------- QKV GEMM
__global__ void __launch_bounds__(128)
gemm_qkv(const bf16_t* __restrict__ xb, const bf16_t* __restrict__ wb,
         const float* __restrict__ bias,
         bf16_t* __restrict__ Qh, bf16_t* __restrict__ Kh, bf16_t* __restrict__ Vt) {
    const int lane = threadIdx.x & 31;
    const int wave = threadIdx.x >> 5;
    const int half = lane >> 4;
    const int l16  = lane & 15;
    const int koff = half * 8;

    const int tile = blockIdx.x * 4 + wave;   // 12288 tiles = 256 x 48
    const int tm = tile & 255;
    const int tn = tile >> 8;
    const int row0 = tm * 64;
    const int col0 = tn * 64;

    AccReg acc[4][4];
    #pragma unroll
    for (int i = 0; i < 4; ++i)
        #pragma unroll
        for (int j = 0; j < 4; ++j)
            #pragma unroll
            for (int r = 0; r < 8; ++r) acc[i][j].f[r] = 0.f;

    for (int kb = 0; kb < D_DIM; kb += 32) {
        ABReg A[4], Bm[4];
        #pragma unroll
        for (int i = 0; i < 4; ++i) {
            const bf16_t* p = xb + (size_t)(row0 + i * 16 + l16) * D_DIM + kb + koff;
            A[i].q[0] = *(const u32x4*)p;
            A[i].q[1] = *(const u32x4*)(p + 16);
        }
        #pragma unroll
        for (int j = 0; j < 4; ++j) {
            const bf16_t* p = wb + (size_t)(col0 + j * 16 + l16) * D_DIM + kb + koff;
            Bm[j].q[0] = *(const u32x4*)p;
            Bm[j].q[1] = *(const u32x4*)(p + 16);
        }
        #pragma unroll
        for (int i = 0; i < 4; ++i)
            #pragma unroll
            for (int j = 0; j < 4; ++j)
                acc[i][j].v = __builtin_amdgcn_wmma_f32_16x16x32_bf16(
                    false, A[i].v, false, Bm[j].v, (short)0, acc[i][j].v, false, false);
    }

    #pragma unroll
    for (int j = 0; j < 4; ++j) {
        const int e    = col0 + j * 16 + l16;
        const float bv = bias[e];
        const int sect = e >> 10;       // 0=q 1=k 2=v
        const int ec   = e & 1023;
        const int h    = ec >> 6;
        const int c    = ec & 63;
        #pragma unroll
        for (int i = 0; i < 4; ++i) {
            #pragma unroll
            for (int r = 0; r < 8; ++r) {
                const int row = row0 + i * 16 + r + half * 8;  // n = t*B + b
                const int t  = row >> 4;
                const int b  = row & 15;
                const int bh = b * H_DIM + h;
                const float v = acc[i][j].f[r] + bv;
                if (sect == 0)
                    Qh[(size_t)bh * (T_DIM * HD) + t * HD + c] = (bf16_t)(v * 0.125f);
                else if (sect == 1)
                    Kh[(size_t)bh * (T_DIM * HD) + t * HD + c] = (bf16_t)v;
                else
                    Vt[(size_t)bh * (HD * T_DIM) + c * T_DIM + t] = (bf16_t)v;
            }
        }
    }
}

// ---------------------------------------------------------------- attention
// One wave (= one workgroup) handles 16 queries of one (b,h). K/V tiles are
// staged into LDS by the Tensor Data Mover with double buffering; S^T = K*Q^T
// keeps softmax per-lane and makes the bf16-packed S^T the A-operand of P*V.
__global__ void __launch_bounds__(32)
attn_fwd(const bf16_t* __restrict__ Qh, const bf16_t* __restrict__ Kh,
         const bf16_t* __restrict__ Vt, bf16_t* __restrict__ ctx) {
    __shared__ __align__(16) bf16_t ldsK[2][32 * HD];   // 8 KB
    __shared__ __align__(16) bf16_t ldsV[2][HD * 32];   // 8 KB

    const int lane = threadIdx.x & 31;
    const int half = lane >> 4;
    const int l16  = lane & 15;
    const int koff = half * 8;

    const int unit = blockIdx.x;      // 16384 = 256 heads * 64 q-tiles
    const int bh = unit >> 6;
    const int qt = unit & 63;
    const int q0 = qt * 16;

    const bf16_t* Qp = Qh + (size_t)bh * (T_DIM * HD);
    const bf16_t* Kp = Kh + (size_t)bh * (T_DIM * HD);
    const bf16_t* Vp = Vt + (size_t)bh * (HD * T_DIM);

    // raw LDS byte offsets (flat shared addr truncates to LDS offset)
    const uint32_t offK[2] = { (uint32_t)(uintptr_t)&ldsK[0][0],
                               (uint32_t)(uintptr_t)&ldsK[1][0] };
    const uint32_t offV[2] = { (uint32_t)(uintptr_t)&ldsV[0][0],
                               (uint32_t)(uintptr_t)&ldsV[1][0] };

    // Q as WMMA B-operand (lane column = query), two 32-wide chunks of head dim
    ABReg Bq[2];
    #pragma unroll
    for (int kc = 0; kc < 2; ++kc) {
        const bf16_t* p = Qp + (size_t)(q0 + l16) * HD + kc * 32 + koff;
        Bq[kc].q[0] = *(const u32x4*)p;
        Bq[kc].q[1] = *(const u32x4*)(p + 16);
    }

    AccReg o[4];
    #pragma unroll
    for (int nt = 0; nt < 4; ++nt)
        #pragma unroll
        for (int r = 0; r < 8; ++r) o[nt].f[r] = 0.f;

    float m_i = -3.0e38f;
    float l_i = 0.f;
    const int nb = (q0 + 15) / 32 + 1;   // causal: key blocks of 32

    // prologue: TDM-issue block 0 (K: 32x64 rows stride 64; V: 64x32 stride 1024)
    tdm_load_tile_2d(offK[0], Kp, HD, 32, HD, T_DIM, HD);
    tdm_load_tile_2d(offV[0], Vp, 32, HD, T_DIM, HD, T_DIM);

    for (int jb = 0; jb < nb; ++jb) {
        const int kbase = jb * 32;
        if (jb + 1 < nb) {   // issue next block into the other buffer
            const int nk = kbase + 32;
            tdm_load_tile_2d(offK[(jb + 1) & 1], Kp + (size_t)nk * HD, HD, 32, HD, T_DIM, HD);
            tdm_load_tile_2d(offV[(jb + 1) & 1], Vp + nk, 32, HD, T_DIM, HD, T_DIM);
            __builtin_amdgcn_s_wait_tensorcnt(2);   // oldest pair (block jb) done
        } else {
            __builtin_amdgcn_s_wait_tensorcnt(0);
        }
        const bf16_t* kl = &ldsK[jb & 1][0];
        const bf16_t* vl = &ldsV[jb & 1][0];

        // S^T tiles: [key 16 x query 16], two key tiles (ds_load_b128 from LDS)
        AccReg st[2];
        #pragma unroll
        for (int kt = 0; kt < 2; ++kt) {
            const bf16_t* p0 = kl + (kt * 16 + l16) * HD + koff;
            ABReg Ka0, Ka1;
            Ka0.q[0] = *(const u32x4*)p0;
            Ka0.q[1] = *(const u32x4*)(p0 + 16);
            Ka1.q[0] = *(const u32x4*)(p0 + 32);
            Ka1.q[1] = *(const u32x4*)(p0 + 48);
            v8f z = {0.f, 0.f, 0.f, 0.f, 0.f, 0.f, 0.f, 0.f};
            st[kt].v = __builtin_amdgcn_wmma_f32_16x16x32_bf16(
                false, Ka0.v, false, Bq[0].v, (short)0, z, false, false);
            st[kt].v = __builtin_amdgcn_wmma_f32_16x16x32_bf16(
                false, Ka1.v, false, Bq[1].v, (short)0, st[kt].v, false, false);
        }

        // causal mask: only the final (diagonal) block is partial
        if (jb == nb - 1) {
            const int q = q0 + l16;
            #pragma unroll
            for (int kt = 0; kt < 2; ++kt)
                #pragma unroll
                for (int r = 0; r < 8; ++r) {
                    const int tk = kbase + kt * 16 + r + half * 8;
                    if (tk > q) st[kt].f[r] = -1.0e30f;
                }
        }

        // online softmax: query is the lane column -> per-lane reductions
        float bm = st[0].f[0];
        #pragma unroll
        for (int kt = 0; kt < 2; ++kt)
            #pragma unroll
            for (int r = 0; r < 8; ++r) bm = fmaxf(bm, st[kt].f[r]);
        bm = fmaxf(bm, __shfl_xor(bm, 16, 32));   // merge key halves of same query
        const float m_new = fmaxf(m_i, bm);
        const float alpha = __expf(m_i - m_new);

        float ls = 0.f;
        ABReg Pa;   // exp(S^T) packed: exactly the A-operand layout for P*V
        #pragma unroll
        for (int kt = 0; kt < 2; ++kt)
            #pragma unroll
            for (int r = 0; r < 4; ++r) {
                const float e0 = __expf(st[kt].f[2 * r]     - m_new);
                const float e1 = __expf(st[kt].f[2 * r + 1] - m_new);
                ls += e0 + e1;
                Pa.u[kt * 4 + r] = pack2_bf16(e0, e1);
            }
        l_i = l_i * alpha + ls;   // per-lane partial (halves merged at end)
        m_i = m_new;

        // rescale O (rows = queries in VGPRs) with alpha broadcast per row
        #pragma unroll
        for (int r = 0; r < 8; ++r) {
            const float ar = __shfl(alpha, r + half * 8, 32);
            #pragma unroll
            for (int nt = 0; nt < 4; ++nt) o[nt].f[r] *= ar;
        }

        // O += P * V  (V^T tile in LDS -> contiguous B-operand ds_load_b128)
        #pragma unroll
        for (int nt = 0; nt < 4; ++nt) {
            const bf16_t* p = vl + (nt * 16 + l16) * 32 + koff;
            ABReg Vb;
            Vb.q[0] = *(const u32x4*)p;
            Vb.q[1] = *(const u32x4*)(p + 16);
            o[nt].v = __builtin_amdgcn_wmma_f32_16x16x32_bf16(
                false, Pa.v, false, Vb.v, (short)0, o[nt].v, false, false);
        }
    }

    // finalize: merge l halves, normalize, scatter to ctx[t,b,d] as bf16
    l_i += __shfl_xor(l_i, 16, 32);
    const float inv = 1.0f / l_i;
    const int b = bh >> 4;
    const int h = bh & 15;
    #pragma unroll
    for (int r = 0; r < 8; ++r) {
        const float ir = __shfl(inv, r + half * 8, 32);
        const int t = q0 + r + half * 8;
        #pragma unroll
        for (int nt = 0; nt < 4; ++nt) {
            const int c = nt * 16 + l16;
            ctx[(size_t)(t * B_DIM + b) * D_DIM + h * HD + c] = (bf16_t)(o[nt].f[r] * ir);
        }
    }
}

// ---------------------------------------------------------------- output GEMM
__global__ void __launch_bounds__(128)
gemm_out(const bf16_t* __restrict__ ctx, const bf16_t* __restrict__ wb,
         const float* __restrict__ bias, float* __restrict__ out) {
    const int lane = threadIdx.x & 31;
    const int wave = threadIdx.x >> 5;
    const int half = lane >> 4;
    const int l16  = lane & 15;
    const int koff = half * 8;

    const int tile = blockIdx.x * 4 + wave;   // 4096 tiles = 256 x 16
    const int tm = tile & 255;
    const int tn = tile >> 8;
    const int row0 = tm * 64;
    const int col0 = tn * 64;

    AccReg acc[4][4];
    #pragma unroll
    for (int i = 0; i < 4; ++i)
        #pragma unroll
        for (int j = 0; j < 4; ++j)
            #pragma unroll
            for (int r = 0; r < 8; ++r) acc[i][j].f[r] = 0.f;

    for (int kb = 0; kb < D_DIM; kb += 32) {
        ABReg A[4], Bm[4];
        #pragma unroll
        for (int i = 0; i < 4; ++i) {
            const bf16_t* p = ctx + (size_t)(row0 + i * 16 + l16) * D_DIM + kb + koff;
            A[i].q[0] = *(const u32x4*)p;
            A[i].q[1] = *(const u32x4*)(p + 16);
        }
        #pragma unroll
        for (int j = 0; j < 4; ++j) {
            const bf16_t* p = wb + (size_t)(col0 + j * 16 + l16) * D_DIM + kb + koff;
            Bm[j].q[0] = *(const u32x4*)p;
            Bm[j].q[1] = *(const u32x4*)(p + 16);
        }
        #pragma unroll
        for (int i = 0; i < 4; ++i)
            #pragma unroll
            for (int j = 0; j < 4; ++j)
                acc[i][j].v = __builtin_amdgcn_wmma_f32_16x16x32_bf16(
                    false, A[i].v, false, Bm[j].v, (short)0, acc[i][j].v, false, false);
    }

    #pragma unroll
    for (int j = 0; j < 4; ++j) {
        const int e = col0 + j * 16 + l16;
        const float bv = bias[e];
        #pragma unroll
        for (int i = 0; i < 4; ++i)
            #pragma unroll
            for (int r = 0; r < 8; ++r) {
                const int row = row0 + i * 16 + r + half * 8;
                out[(size_t)row * D_DIM + e] = acc[i][j].f[r] + bv;
            }
    }
}

// ---------------------------------------------------------------- launcher
extern "C" void kernel_launch(void* const* d_in, const int* in_sizes, int n_in,
                              void* d_out, int out_size, void* d_ws, size_t ws_size,
                              hipStream_t stream) {
    (void)in_sizes; (void)n_in; (void)out_size; (void)ws_size;
    const float* x     = (const float*)d_in[0];
    const float* w_in  = (const float*)d_in[1];
    const float* b_in  = (const float*)d_in[2];
    const float* w_out = (const float*)d_in[3];
    const float* b_out = (const float*)d_in[4];
    float* out = (float*)d_out;

    char* ws = (char*)d_ws;
    size_t off = 0;
    bf16_t* xb    = (bf16_t*)(ws + off); off += (size_t)NROW * D_DIM * 2;          // 32 MB
    bf16_t* winb  = (bf16_t*)(ws + off); off += (size_t)E3D * D_DIM * 2;           //  6 MB
    bf16_t* woutb = (bf16_t*)(ws + off); off += (size_t)D_DIM * D_DIM * 2;         //  2 MB
    bf16_t* Qhp   = (bf16_t*)(ws + off); off += (size_t)B_DIM * H_DIM * T_DIM * HD * 2; // 32 MB
    bf16_t* Khp   = (bf16_t*)(ws + off); off += (size_t)B_DIM * H_DIM * T_DIM * HD * 2; // 32 MB
    bf16_t* Vtp   = (bf16_t*)(ws + off); off += (size_t)B_DIM * H_DIM * HD * T_DIM * 2; // 32 MB
    bf16_t* ctxb  = (bf16_t*)(ws + off); off += (size_t)NROW * D_DIM * 2;          // 32 MB

    cvt_f32_bf16<<<4096, 256, 0, stream>>>(x,     xb,    NROW * D_DIM);
    cvt_f32_bf16<<<1024, 256, 0, stream>>>(w_in,  winb,  E3D * D_DIM);
    cvt_f32_bf16<<< 512, 256, 0, stream>>>(w_out, woutb, D_DIM * D_DIM);
    gemm_qkv<<<3072, 128, 0, stream>>>(xb, winb, b_in, Qhp, Khp, Vtp);
    attn_fwd<<<16384, 32, 0, stream>>>(Qhp, Khp, Vtp, ctxb);
    gemm_out<<<1024, 128, 0, stream>>>(ctxb, woutb, b_out, out);
}
